// Element_13219909337211
// MI455X (gfx1250) — matrix operand order repaired and checked by reading, not compiled, not run on hardware
//
#include <hip/hip_runtime.h>
#include <hip/hip_bf16.h>

// ---------------- problem constants ----------------
#define BATCH 8
#define SEQ   2048
#define DIM   512
#define DIN   1024            // N*D concat width
#define MROWS (BATCH * SEQ)   // 16384
#define QSCALE 0.044194173824159216f  // 512^-0.5

typedef __attribute__((ext_vector_type(16))) __bf16 v16bf;
typedef __attribute__((ext_vector_type(8)))  __bf16 v8bf;
typedef __attribute__((ext_vector_type(8)))  float  v8f;

#define WMMA_BF16 __builtin_amdgcn_wmma_f32_16x16x32_bf16

// ---------------- WMMA fragment loaders ----------------
// A fragment (16x32, bf16). base -> row 0 of the 16-row tile, row-major, leading dim ld.
// lane: m = lane&15, half h = lane>>4.  element e: K = e + 16*h + k0
__device__ __forceinline__ v16bf load_frag_a(const __bf16* base, int ld, int k0) {
  const int lane = threadIdx.x & 31;
  const int m = lane & 15;
  const int h = lane >> 4;
  const __bf16* p = base + (size_t)m * ld + k0 + h * 16;
  v16bf a = *(const v16bf*)(p);       // K = k0+16h .. +15
  return a;
}

// B fragment (32x16, bf16) where the logical B[k][n] is stored as src[n][k]
// (row-major by n, leading dim ld). lane: n = lane&15, K = e + 16*(lane>>4) + k0.
__device__ __forceinline__ v16bf load_frag_b(const __bf16* base, int ld, int k0) {
  const int lane = threadIdx.x & 31;
  const int n = lane & 15;
  const int h = lane >> 4;
  const __bf16* p = base + (size_t)n * ld + k0 + h * 16;
  return *(const v16bf*)p;            // 16 contiguous bf16
}

// ---------------- stage 1: convert / pack ----------------
__global__ void concat_convert_kernel(const float* __restrict__ x0,
                                      const float* __restrict__ x1,
                                      __bf16* __restrict__ xb) {
  size_t i = (size_t)blockIdx.x * blockDim.x + threadIdx.x;   // over MROWS*DIN
  if (i >= (size_t)MROWS * DIN) return;
  size_t row = i >> 10;
  int    c   = (int)(i & 1023);
  float  v   = (c < DIM) ? x0[row * DIM + c] : x1[row * DIM + (c - DIM)];
  xb[i] = (__bf16)v;
}

// W [DIN, DIM] fp32 -> Wt [DIM, DIN] bf16 (transposed so B-frags are contiguous in K)
__global__ void wtrans_convert_kernel(const float* __restrict__ W,
                                      __bf16* __restrict__ Wt) {
  int i = blockIdx.x * blockDim.x + threadIdx.x;              // over DIM*DIN
  if (i >= DIM * DIN) return;
  int nrow = i >> 10;        // n: 0..511
  int kcol = i & 1023;       // k: 0..1023
  Wt[i] = (__bf16)W[(size_t)kcol * DIM + nrow];
}

// ---------------- stage 2: QKV projection GEMM ----------------
// C[m,n] = sum_k X[m,k] * Wt[n,k]; per wave: 32x64 tile (2 A-frags x 4 B-frags,
// 8 accumulators), software-pipelined ping-pong on the K loop.
// vtrans==0: out row-major bf16 [MROWS, DIM] (q is pre-scaled by QSCALE)
// vtrans==1: out = Vt bf16 [BATCH][DIM][SEQ]
__global__ void __launch_bounds__(256, 1) gemm_qkv_kernel(const __bf16* __restrict__ X,
                                                          const __bf16* __restrict__ Wt,
                                                          __bf16* __restrict__ out,
                                                          float scale, int vtrans) {
  const int wave = (int)((blockIdx.x * blockDim.x + threadIdx.x) >> 5);
  const int mi = wave >> 3;          // 0..511  (row tile of 32)
  const int n0 = (wave & 7) * 64;    // column group
  const int lane = threadIdx.x & 31;
  const int n = lane & 15, h = lane >> 4;

  v8f acc[2][4] = {};
  const __bf16* arow0 = X + (size_t)(mi * 32) * DIN;
  const __bf16* arow1 = arow0 + (size_t)16 * DIN;

  // prime the pipeline
  v16bf a0 = load_frag_a(arow0, DIN, 0);
  v16bf a1 = load_frag_a(arow1, DIN, 0);
  v16bf b0 = load_frag_b(Wt + (size_t)(n0 +  0) * DIN, DIN, 0);
  v16bf b1 = load_frag_b(Wt + (size_t)(n0 + 16) * DIN, DIN, 0);
  v16bf b2 = load_frag_b(Wt + (size_t)(n0 + 32) * DIN, DIN, 0);
  v16bf b3 = load_frag_b(Wt + (size_t)(n0 + 48) * DIN, DIN, 0);

  for (int k0 = 0; k0 < DIN; k0 += 32) {
    const int kn = (k0 + 32 < DIN) ? k0 + 32 : 0;   // wrap: last prefetch harmless
    // issue next-step loads BEFORE this step's wmmas (ping-pong)
    v16bf na0 = load_frag_a(arow0, DIN, kn);
    v16bf na1 = load_frag_a(arow1, DIN, kn);
    v16bf nb0 = load_frag_b(Wt + (size_t)(n0 +  0) * DIN, DIN, kn);
    v16bf nb1 = load_frag_b(Wt + (size_t)(n0 + 16) * DIN, DIN, kn);
    v16bf nb2 = load_frag_b(Wt + (size_t)(n0 + 32) * DIN, DIN, kn);
    v16bf nb3 = load_frag_b(Wt + (size_t)(n0 + 48) * DIN, DIN, kn);

    acc[0][0] = WMMA_BF16(false, a0, false, b0, (short)0, acc[0][0], false, false);
    acc[0][1] = WMMA_BF16(false, a0, false, b1, (short)0, acc[0][1], false, false);
    acc[0][2] = WMMA_BF16(false, a0, false, b2, (short)0, acc[0][2], false, false);
    acc[0][3] = WMMA_BF16(false, a0, false, b3, (short)0, acc[0][3], false, false);
    acc[1][0] = WMMA_BF16(false, a1, false, b0, (short)0, acc[1][0], false, false);
    acc[1][1] = WMMA_BF16(false, a1, false, b1, (short)0, acc[1][1], false, false);
    acc[1][2] = WMMA_BF16(false, a1, false, b2, (short)0, acc[1][2], false, false);
    acc[1][3] = WMMA_BF16(false, a1, false, b3, (short)0, acc[1][3], false, false);

    a0 = na0; a1 = na1; b0 = nb0; b1 = nb1; b2 = nb2; b3 = nb3;
  }

  if (!vtrans) {
#pragma unroll
    for (int t = 0; t < 2; ++t)
#pragma unroll
      for (int j = 0; j < 4; ++j)
#pragma unroll
        for (int r = 0; r < 8; ++r) {
          int m = mi * 32 + t * 16 + r + 8 * h;
          out[(size_t)m * DIM + n0 + j * 16 + n] = (__bf16)(acc[t][j][r] * scale);
        }
  } else {
#pragma unroll
    for (int t = 0; t < 2; ++t)
#pragma unroll
      for (int j = 0; j < 4; ++j)
#pragma unroll
        for (int r = 0; r < 8; ++r) {
          int m  = mi * 32 + t * 16 + r + 8 * h;
          int bb = m >> 11;            // /SEQ
          int tt = m & (SEQ - 1);
          int d  = n0 + j * 16 + n;
          out[((size_t)bb * DIM + d) * SEQ + tt] = (__bf16)(acc[t][j][r] * scale);
        }
  }
}

// ---------------- stage 3: fused causal flash attention ----------------
// block: 256 threads (8 waves). Each block: batch blockIdx.y, query rows [t0, t0+64).
// Each wave owns a 64-wide slice of the output dim (d0 = wave*64) -> 16 v8f accumulators.
__global__ void __launch_bounds__(256, 1) attn_kernel(const __bf16* __restrict__ q,
                                                      const __bf16* __restrict__ k,
                                                      const __bf16* __restrict__ vt,
                                                      float* __restrict__ out) {
  const int t0   = blockIdx.x * 64;
  const int b    = blockIdx.y;
  const int tid  = threadIdx.x;
  const int wave = tid >> 5;
  const int lane = tid & 31;
  const int n = lane & 15, h = lane >> 4;

  __shared__ __align__(64) float  Sbuf[64][64];   // raw scores (C layout dump)
  __shared__ __align__(64) __bf16 Pbuf[64][64];   // probabilities (A-frag source)
  __shared__ float mI[64], lI[64], aI[64];
  __shared__ float pmax[4][64], psum[4][64];      // 4-way parallel softmax partials

  if (tid < 64) { mI[tid] = -1e30f; lI[tid] = 0.f; }

  v8f acc[4][4] = {};                 // [m-tile][n-tile] of this wave's O slice
  const int d0 = wave * 64;
  const __bf16* qb = q  + (size_t)b * SEQ * DIM;
  const __bf16* kb = k  + (size_t)b * SEQ * DIM;
  const __bf16* vb = vt + (size_t)b * DIM * SEQ;

  const int srow = tid & 63;          // softmax row owned by this thread
  const int sseg = tid >> 6;          // softmax 16-col segment (0..3)

  for (int s0 = 0; s0 <= t0; s0 += 64) {
    __syncthreads();
    // prefetch next key block (K rows + V rows) into cache while we compute
    if (s0 + 64 <= t0) {
      const __bf16* pk = kb + (size_t)(s0 + 64 + (tid >> 2)) * DIM + (tid & 3) * 128;
      __builtin_prefetch(pk, 0, 1);
      const __bf16* pv0 = vb + (size_t)(tid * 2) * SEQ + s0 + 64;
      const __bf16* pv1 = vb + (size_t)(tid * 2 + 1) * SEQ + s0 + 64;
      __builtin_prefetch(pv0, 0, 1);
      __builtin_prefetch(pv1, 0, 1);
    }
    // ---- score phase: 16 tiles (64x64), 2 per wave; S = q (scaled) . k^T ----
#pragma unroll
    for (int tt = 0; tt < 2; ++tt) {
      const int tile = wave * 2 + tt;
      const int smi = tile >> 2, snj = tile & 3;
      v8f s = {};
      const __bf16* arow = qb + (size_t)(t0 + smi * 16) * DIM;
      const __bf16* brow = kb + (size_t)(s0 + snj * 16) * DIM;
      v16bf a  = load_frag_a(arow, DIM, 0);
      v16bf bf = load_frag_b(brow, DIM, 0);
      for (int k0 = 0; k0 < DIM; k0 += 32) {
        const int kn = (k0 + 32 < DIM) ? k0 + 32 : 0;
        v16bf an = load_frag_a(arow, DIM, kn);   // issued before the wmma
        v16bf bn = load_frag_b(brow, DIM, kn);
        s = WMMA_BF16(false, a, false, bf, (short)0, s, false, false);
        a = an; bf = bn;
      }
#pragma unroll
      for (int r = 0; r < 8; ++r)
        Sbuf[smi * 16 + r + 8 * h][snj * 16 + n] = s[r];
    }
    __syncthreads();
    // ---- online softmax, 4-way parallel per row (all 256 threads) ----
    {   // phase 1: partial max over this thread's 16-column segment
      const int gr = t0 + srow;
      float pm = -1e30f;
#pragma unroll
      for (int jj = 0; jj < 16; ++jj) {
        const int j = sseg * 16 + jj;
        float sv = (s0 + j <= gr) ? Sbuf[srow][j] : -1e30f;
        pm = fmaxf(pm, sv);
      }
      pmax[sseg][srow] = pm;
    }
    __syncthreads();
    {   // phase 2: exp + partial sums (mNew recomputed redundantly, no extra sync)
      const int gr = t0 + srow;
      const float mOld = mI[srow];
      const float mNew = fmaxf(mOld, fmaxf(fmaxf(pmax[0][srow], pmax[1][srow]),
                                           fmaxf(pmax[2][srow], pmax[3][srow])));
      float sum = 0.f;
#pragma unroll
      for (int jj = 0; jj < 16; ++jj) {
        const int j = sseg * 16 + jj;
        float sv = (s0 + j <= gr) ? Sbuf[srow][j] : -1e30f;
        float p  = __expf(sv - mNew);
        sum += p;
        Pbuf[srow][j] = (__bf16)p;
      }
      psum[sseg][srow] = sum;
    }
    __syncthreads();
    if (tid < 64) {   // phase 3: fold partials into running stats
      const float mOld = mI[tid];
      const float mNew = fmaxf(mOld, fmaxf(fmaxf(pmax[0][tid], pmax[1][tid]),
                                           fmaxf(pmax[2][tid], pmax[3][tid])));
      const float alpha = __expf(mOld - mNew);
      const float sum = psum[0][tid] + psum[1][tid] + psum[2][tid] + psum[3][tid];
      mI[tid] = mNew;
      lI[tid] = lI[tid] * alpha + sum;
      aI[tid] = alpha;
    }
    __syncthreads();
    // ---- PV phase: O = O*alpha + P @ V  (this wave's 64 columns) ----
    // V fragments are independent of the m-tile: load the 8 frags ONCE per block.
    v16bf vf[2][4];
#pragma unroll
    for (int kk = 0; kk < 2; ++kk)
#pragma unroll
      for (int nj = 0; nj < 4; ++nj)
        vf[kk][nj] = load_frag_b(vb + (size_t)(d0 + nj * 16) * SEQ + s0, SEQ, kk * 32);
#pragma unroll
    for (int mi2 = 0; mi2 < 4; ++mi2) {
      float ar[8];                    // hoist per-row alphas out of the nj loop
#pragma unroll
      for (int r = 0; r < 8; ++r) ar[r] = aI[mi2 * 16 + r + 8 * h];
#pragma unroll
      for (int nj = 0; nj < 4; ++nj)
#pragma unroll
        for (int r = 0; r < 8; ++r)
          acc[mi2][nj][r] *= ar[r];
#pragma unroll
      for (int kk = 0; kk < 2; ++kk) {
        v16bf a = load_frag_a(&Pbuf[mi2 * 16][0], 64, kk * 32);
#pragma unroll
        for (int nj = 0; nj < 4; ++nj)
          acc[mi2][nj] = WMMA_BF16(false, a, false, vf[kk][nj], (short)0,
                                   acc[mi2][nj], false, false);
      }
    }
  }
  __syncthreads();
  // ---- epilogue: divide by softmax denom, write fp32 ----
#pragma unroll
  for (int mi2 = 0; mi2 < 4; ++mi2) {
    float inv[8];                     // hoist reciprocals out of the nj loop
#pragma unroll
    for (int r = 0; r < 8; ++r) inv[r] = 1.f / lI[mi2 * 16 + r + 8 * h];
#pragma unroll
    for (int nj = 0; nj < 4; ++nj)
#pragma unroll
      for (int r = 0; r < 8; ++r) {
        const int row = mi2 * 16 + r + 8 * h;
        out[((size_t)b * SEQ + t0 + row) * DIM + d0 + nj * 16 + n] =
            acc[mi2][nj][r] * inv[r];
      }
  }
}

// ---------------- stage 4: LayerNorm (in-place on d_out) ----------------
__global__ void __launch_bounds__(128) ln_kernel(float* __restrict__ out,
                                                 const float* __restrict__ gamma,
                                                 const float* __restrict__ beta) {
  const int row = blockIdx.x;
  const int tid = threadIdx.x;          // 128 threads, 4 elems each
  float* p = out + (size_t)row * DIM;
  float4 v = ((const float4*)p)[tid];
  __shared__ float red[128], red2[128];
  red[tid]  = v.x + v.y + v.z + v.w;
  red2[tid] = v.x * v.x + v.y * v.y + v.z * v.z + v.w * v.w;
  __syncthreads();
  for (int off = 64; off > 0; off >>= 1) {
    if (tid < off) { red[tid] += red[tid + off]; red2[tid] += red2[tid + off]; }
    __syncthreads();
  }
  const float mean = red[0] * (1.0f / DIM);
  const float var  = red2[0] * (1.0f / DIM) - mean * mean;
  const float inv  = rsqrtf(var + 1e-5f);
  float4 g = ((const float4*)gamma)[tid];
  float4 bb = ((const float4*)beta)[tid];
  float4 o;
  o.x = (v.x - mean) * inv * g.x + bb.x;
  o.y = (v.y - mean) * inv * g.y + bb.y;
  o.z = (v.z - mean) * inv * g.z + bb.z;
  o.w = (v.w - mean) * inv * g.w + bb.w;
  ((float4*)p)[tid] = o;
}

// ---------------- host launcher ----------------
extern "C" void kernel_launch(void* const* d_in, const int* in_sizes, int n_in,
                              void* d_out, int out_size, void* d_ws, size_t ws_size,
                              hipStream_t stream) {
  const float* x0    = (const float*)d_in[0];
  const float* x1    = (const float*)d_in[1];
  const float* Wq    = (const float*)d_in[2];
  const float* Wk    = (const float*)d_in[3];
  const float* Wv    = (const float*)d_in[4];
  const float* gamma = (const float*)d_in[5];
  const float* beta  = (const float*)d_in[6];
  float* out = (float*)d_out;

  // workspace layout (bytes)
  char* ws = (char*)d_ws;
  __bf16* xb  = (__bf16*)(ws);                         // 16384*1024*2 = 32 MiB
  __bf16* wtq = (__bf16*)(ws + 33554432);              // 512*1024*2 each
  __bf16* wtk = wtq + (size_t)DIM * DIN;
  __bf16* wtv = wtk + (size_t)DIM * DIN;
  __bf16* qb  = (__bf16*)(ws + 36700160);              // 16384*512*2 each
  __bf16* kb  = qb + (size_t)MROWS * DIM;
  __bf16* vtb = kb + (size_t)MROWS * DIM;              // [BATCH][DIM][SEQ]

  // 1) pack x = concat(x0,x1) -> bf16
  concat_convert_kernel<<<(MROWS * (size_t)DIN + 255) / 256, 256, 0, stream>>>(x0, x1, xb);
  // 2) transpose+convert weights
  wtrans_convert_kernel<<<(DIM * DIN + 255) / 256, 256, 0, stream>>>(Wq, wtq);
  wtrans_convert_kernel<<<(DIM * DIN + 255) / 256, 256, 0, stream>>>(Wk, wtk);
  wtrans_convert_kernel<<<(DIM * DIN + 255) / 256, 256, 0, stream>>>(Wv, wtv);
  // 3) QKV projections (4096 waves = 512 blocks x 8 waves, 32x64 tile/wave)
  gemm_qkv_kernel<<<512, 256, 0, stream>>>(xb, wtq, qb, QSCALE, 0);  // q pre-scaled
  gemm_qkv_kernel<<<512, 256, 0, stream>>>(xb, wtk, kb, 1.0f, 0);
  gemm_qkv_kernel<<<512, 256, 0, stream>>>(xb, wtv, vtb, 1.0f, 1);  // v transposed
  // 4) fused causal attention
  attn_kernel<<<dim3(SEQ / 64, BATCH), 256, 0, stream>>>(qb, kb, vtb, out);
  // 5) LayerNorm in place
  ln_kernel<<<BATCH * SEQ, 128, 0, stream>>>(out, gamma, beta);
}
